// UWMRMoE_34720515621036
// MI455X (gfx1250) — compile-verified, matrix operand order checked by reading, not run on hardware
//
#include <hip/hip_runtime.h>
#include <math.h>

#define N_TOK 16384
#define D_DIM 2560
#define M_DIM 1024
#define E_NUM 8
#define LOAD_COEFF 0.01f
#define BPITCH 40   // LDS row pitch in bf16 elements (32 data + 8 pad, 80B = 16B-aligned)

typedef __bf16 bf16_t;
typedef __attribute__((ext_vector_type(16))) __bf16 v16bf;
typedef __attribute__((ext_vector_type(8)))  __bf16 v8bf;
typedef __attribute__((ext_vector_type(8)))  float  v8f;

struct BPair { v8bf lo, hi; };

__device__ __forceinline__ bf16_t f2bf(float f) {
  unsigned u = __builtin_bit_cast(unsigned, f);
  u += 0x7fffu + ((u >> 16) & 1u);           // round-to-nearest-even
  unsigned short h = (unsigned short)(u >> 16);
  return __builtin_bit_cast(bf16_t, h);
}

// Per-lane fragment load for 16-bit 16x32 A/B WMMA layout (global memory).
__device__ __forceinline__ v16bf load_frag(const bf16_t* __restrict__ rowp, int k0) {
  BPair t;
  t.lo = *(const v8bf*)(rowp + k0);        // K = k0..k0+7
  t.hi = *(const v8bf*)(rowp + k0 + 16);   // K = k0+16..k0+23
  return __builtin_bit_cast(v16bf, t);
}

// Fragment load from an LDS tile: row mrel (0..63), k0rel = (lane>=16 ? 8 : 0).
__device__ __forceinline__ v16bf lds_frag(const bf16_t* base, int mrel, int k0rel) {
  const bf16_t* p = base + mrel * BPITCH + k0rel;
  BPair t;
  t.lo = *(const v8bf*)p;
  t.hi = *(const v8bf*)(p + 16);
  return __builtin_bit_cast(v16bf, t);
}

__device__ __forceinline__ v8f wmma_bf16(v16bf a, v16bf b, v8f c) {
  return __builtin_amdgcn_wmma_f32_16x16x32_bf16(false, a, false, b, (short)0, c, false, false);
}

// Async copy: 16B per lane from global to LDS (ASYNCcnt-tracked DMA path).
__device__ __forceinline__ void async_copy_b128(const bf16_t* gsrc, bf16_t* ldst) {
  unsigned lds_off = (unsigned)(unsigned long long)(const void*)ldst;  // LDS aperture: addr[31:0]
  asm volatile("global_load_async_to_lds_b128 %0, %1, off"
               :: "v"(lds_off), "v"((unsigned long long)gsrc)
               : "memory");
}
#define WAIT_ASYNC(n) asm volatile("s_wait_asynccnt " #n ::: "memory")

// ---------------------------------------------------------------- conversions
__global__ __launch_bounds__(256) void cvt_x_kernel(const float* __restrict__ in,
                                                    bf16_t* __restrict__ out) {
  size_t i = ((size_t)blockIdx.x * 256 + threadIdx.x) * 4;
#pragma unroll
  for (int k = 0; k < 4; ++k) out[i + k] = f2bf(in[i + k]);
}

// in: [Z][R][C] f32 row-major -> out: [Z][C][R] bf16 (K-contiguous for WMMA B)
__global__ __launch_bounds__(256) void transpose_cvt(const float* __restrict__ in,
                                                     bf16_t* __restrict__ out,
                                                     int R, int C) {
  __shared__ float tile[32][33];
  int z = blockIdx.z;
  const float* inz = in + (size_t)z * R * C;
  bf16_t* outz = out + (size_t)z * R * C;
  int c0 = blockIdx.x * 32, r0 = blockIdx.y * 32;
  int tx = threadIdx.x, ty = threadIdx.y;  // 32 x 8
#pragma unroll
  for (int i = 0; i < 32; i += 8)
    tile[ty + i][tx] = inz[(size_t)(r0 + ty + i) * C + (c0 + tx)];
  __syncthreads();
#pragma unroll
  for (int i = 0; i < 32; i += 8)
    outz[(size_t)(c0 + ty + i) * R + (r0 + tx)] = f2bf(tile[tx][ty + i]);
}

// ---------------------------------------------------------------- router
__global__ __launch_bounds__(256) void router_kernel(
    const float* __restrict__ x, const float* __restrict__ U,
    const float* __restrict__ rw, const float* __restrict__ sb,
    const float* __restrict__ gb, int* __restrict__ tok_e,
    float* __restrict__ tok_w, float* __restrict__ partials) {
  __shared__ float pw[8][E_NUM];
  int lane = threadIdx.x & 31, wid = threadIdx.x >> 5;
  int t = blockIdx.x * 8 + wid;
  float acc[E_NUM];
#pragma unroll
  for (int e = 0; e < E_NUM; ++e) acc[e] = 0.f;
  const float* xr = x + (size_t)t * D_DIM;
  for (int d = lane; d < D_DIM; d += 32) {
    float xv = xr[d];
    const float* r = rw + (size_t)d * E_NUM;
#pragma unroll
    for (int e = 0; e < E_NUM; ++e) acc[e] += xv * r[e];
  }
#pragma unroll
  for (int e = 0; e < E_NUM; ++e)
    for (int o = 16; o > 0; o >>= 1) acc[e] += __shfl_down(acc[e], o, 32);
  if (lane == 0) {
    float u = fminf(fmaxf(U[t], 0.f), 1.f);
    float l[E_NUM], m = -1e30f;
#pragma unroll
    for (int e = 0; e < E_NUM; ++e) {
      l[e] = acc[e] + u * sb[e] + (1.f - u) * gb[e];
      m = fmaxf(m, l[e]);
    }
    float p[E_NUM], s = 0.f;
#pragma unroll
    for (int e = 0; e < E_NUM; ++e) { p[e] = __expf(l[e] - m); s += p[e]; }
    float inv = 1.f / s;
#pragma unroll
    for (int e = 0; e < E_NUM; ++e) p[e] *= inv;
    int e0 = 0;
#pragma unroll
    for (int e = 1; e < E_NUM; ++e) if (p[e] > p[e0]) e0 = e;
    int e1 = (e0 == 0) ? 1 : 0;
#pragma unroll
    for (int e = 0; e < E_NUM; ++e)
      if (e != e0 && p[e] > p[e1]) e1 = e;
    tok_e[2 * t] = e0; tok_e[2 * t + 1] = e1;
    tok_w[2 * t] = p[e0]; tok_w[2 * t + 1] = p[e1];
#pragma unroll
    for (int e = 0; e < E_NUM; ++e) pw[wid][e] = p[e];
  }
  __syncthreads();
  if (threadIdx.x < E_NUM) {
    float s = 0.f;
    for (int w = 0; w < 8; ++w) s += pw[w][threadIdx.x];  // fixed order
    partials[(size_t)blockIdx.x * E_NUM + threadIdx.x] = s;
  }
}

__global__ void balloss_kernel(const float* __restrict__ partials, int nblocks,
                               float* __restrict__ loss_out) {
  __shared__ float mp[E_NUM];
  if (threadIdx.x < E_NUM) {
    float s = 0.f;
    for (int b = 0; b < nblocks; ++b) s += partials[(size_t)b * E_NUM + threadIdx.x];
    mp[threadIdx.x] = s / (float)N_TOK;
  }
  __syncthreads();
  if (threadIdx.x == 0) {
    float t = 0.f;
    for (int e = 0; e < E_NUM; ++e) t += mp[e] * mp[e];
    *loss_out = (float)E_NUM * t * LOAD_COEFF;
  }
}

// ---------------------------------------------------------------- buckets
__global__ void zero_counts_kernel(int* counts) {
  if (threadIdx.x < E_NUM) counts[threadIdx.x] = 0;
}

__global__ __launch_bounds__(256) void count_kernel(const int* __restrict__ tok_e,
                                                    int* __restrict__ counts) {
  int t = blockIdx.x * 256 + threadIdx.x;
  if (t < N_TOK) {
    atomicAdd(&counts[tok_e[2 * t]], 1);
    atomicAdd(&counts[tok_e[2 * t + 1]], 1);
  }
}

__global__ void scan_kernel(const int* __restrict__ counts, int* __restrict__ offs) {
  if (threadIdx.x == 0) {
    int s = 0;
    for (int e = 0; e < E_NUM; ++e) { offs[e] = s; s += counts[e]; }
    offs[E_NUM] = s;
  }
}

// One wave per expert; bucket entries in token-id order => deterministic.
__global__ __launch_bounds__(32) void place_kernel(
    const int* __restrict__ tok_e, const float* __restrict__ tok_w,
    const int* __restrict__ offs, int* __restrict__ btok, float* __restrict__ bw) {
  int e = blockIdx.x;
  int lane = threadIdx.x;
  int base = offs[e];
  int cur = 0;
  for (int t0 = 0; t0 < N_TOK; t0 += 32) {
    int t = t0 + lane;
    int slot = -1;
    if (tok_e[2 * t] == e) slot = 0;
    else if (tok_e[2 * t + 1] == e) slot = 1;
    unsigned long long mask = __ballot(slot >= 0);
    if (slot >= 0) {
      int rank = __popcll(mask & ((1ull << lane) - 1ull));
      btok[base + cur + rank] = t;
      bw[base + cur + rank] = tok_w[2 * t + slot];
    }
    cur += __popcll(mask);
  }
}

__global__ __launch_bounds__(256) void zero_out_kernel(float* __restrict__ out, size_t n) {
  size_t i = (size_t)blockIdx.x * blockDim.x + threadIdx.x;
  size_t st = (size_t)gridDim.x * blockDim.x;
  for (; i < n; i += st) out[i] = 0.f;
}

// ------------------------------------------------ stage 1: h = silu(x@G)*(x@Up)
// Block tile 128 rows x 64 cols, 8 waves (4x2), wave tile 32x32 (2x2 WMMA tiles).
// Weight tiles (gate & up, 64x32 bf16) staged in LDS via async-to-LDS DMA,
// double-buffered; each wave issues 2 b128 async copies per K-step.
__global__ __launch_bounds__(256) void gemm_gateup(
    const bf16_t* __restrict__ xb,   // [N][D] bf16
    const bf16_t* __restrict__ gT,   // [Eg][M][D] bf16 (gate^T)
    const bf16_t* __restrict__ uT,   // [Eg][M][D] bf16 (up^T)
    bf16_t* __restrict__ hout,       // rows x M bf16
    const int* __restrict__ btok,    // null => identity (shared expert)
    const int* __restrict__ counts,  // null => N_TOK
    const int* __restrict__ offs) {  // null => 0
  __shared__ bf16_t lg[2][64 * BPITCH];
  __shared__ bf16_t lu[2][64 * BPITCH];
  int e = blockIdx.z;
  int cnt = counts ? counts[e] : N_TOK;
  int rb = blockIdx.y * 128;
  if (rb >= cnt) return;                 // uniform per block
  int off = offs ? offs[e] : 0;
  const bf16_t* g = gT + (size_t)e * M_DIM * D_DIM;
  const bf16_t* u = uT + (size_t)e * M_DIM * D_DIM;
  int lane = threadIdx.x & 31, wid = threadIdx.x >> 5;
  int wr = wid & 3, wc = wid >> 2;
  int half = lane >> 4, ln = lane & 15;
  int rowbase = rb + 32 * wr;
  int colblock = blockIdx.x * 64;
  int colbase = colblock + 32 * wc;

  // A row pointers (gathered tokens)
  const bf16_t* arow[2];
#pragma unroll
  for (int rs = 0; rs < 2; ++rs) {
    int r = rowbase + 16 * rs + ln;
    int rr = (r < cnt) ? r : (cnt - 1);
    int tok = btok ? btok[off + rr] : rr;
    arow[rs] = xb + (size_t)tok * D_DIM;
  }

  // Async copy assignment: wave handles one matrix (wid&1), a 16-row group
  // (wid>>1); each lane copies 16B; 2 async instrs per wave per K-step.
  int cmat = wid & 1;
  int crow = (wid >> 1) * 16 + (lane >> 2);   // rows crow and crow+8
  int cchunk = (lane & 3) * 8;                // element offset in row (16B chunks)
  const bf16_t* csrc = (cmat ? u : g) + (size_t)(colblock + crow) * D_DIM + cchunk;
  bf16_t* cdst0 = (cmat ? lu[0] : lg[0]) + crow * BPITCH + cchunk;
  bf16_t* cdst1 = (cmat ? lu[1] : lg[1]) + crow * BPITCH + cchunk;

#define S1_ISSUE(kk_, b_)                                                      \
  do {                                                                         \
    bf16_t* d_ = (b_) ? cdst1 : cdst0;                                         \
    async_copy_b128(csrc + (kk_), d_);                                         \
    async_copy_b128(csrc + (size_t)8 * D_DIM + (kk_), d_ + 8 * BPITCH);        \
  } while (0)

  v8f accg[2][2], accu[2][2];
  v8f zv = {};
#pragma unroll
  for (int rs = 0; rs < 2; ++rs)
#pragma unroll
    for (int cs = 0; cs < 2; ++cs) { accg[rs][cs] = zv; accu[rs][cs] = zv; }

  S1_ISSUE(0, 0);
  int buf = 0;
  for (int kk = 0; kk < D_DIM; kk += 32) {
    if (kk + 32 < D_DIM) {
      S1_ISSUE(kk + 32, buf ^ 1);
      WAIT_ASYNC(2);                 // current buffer's 2 copies complete
    } else {
      WAIT_ASYNC(0);
    }
    __syncthreads();                 // all waves' copies visible

    int k0 = kk + half * 8;
    if (kk + 256 < D_DIM) __builtin_prefetch(arow[0] + kk + 256, 0, 1);
    const bf16_t* tg = lg[buf];
    const bf16_t* tu = lu[buf];
    v16bf a[2], Bg[2], Bu[2];
#pragma unroll
    for (int rs = 0; rs < 2; ++rs) a[rs] = load_frag(arow[rs], k0);
#pragma unroll
    for (int cs = 0; cs < 2; ++cs) {
      int mrel = 32 * wc + 16 * cs + ln;
      Bg[cs] = lds_frag(tg, mrel, half * 8);
      Bu[cs] = lds_frag(tu, mrel, half * 8);
    }
#pragma unroll
    for (int rs = 0; rs < 2; ++rs)
#pragma unroll
      for (int cs = 0; cs < 2; ++cs) {
        accg[rs][cs] = wmma_bf16(a[rs], Bg[cs], accg[rs][cs]);
        accu[rs][cs] = wmma_bf16(a[rs], Bu[cs], accu[rs][cs]);
      }
    __syncthreads();                 // reads done before next overwrite
    buf ^= 1;
  }
#undef S1_ISSUE

#pragma unroll
  for (int rs = 0; rs < 2; ++rs)
#pragma unroll
    for (int cs = 0; cs < 2; ++cs) {
      int col = colbase + 16 * cs + ln;
#pragma unroll
      for (int j = 0; j < 8; ++j) {
        int row = rowbase + 16 * rs + j + 8 * half;
        if (row < cnt) {
          float gv = accg[rs][cs][j], uv = accu[rs][cs][j];
          float hv = gv * __builtin_amdgcn_rcpf(1.f + __expf(-gv)) * uv;
          hout[(size_t)(off + row) * M_DIM + col] = f2bf(hv);
        }
      }
    }
}

// ------------------------------------------------ stage 2: out += w * (h @ down)
// down^T tile (64x32 bf16) staged in LDS via async DMA; 1 async instr/wave/step.
__global__ __launch_bounds__(256) void gemm_down(
    const bf16_t* __restrict__ hbase,  // rows x M bf16
    const bf16_t* __restrict__ dT,     // [Eg][D][M] bf16 (down^T)
    float* __restrict__ out,           // [N][D] f32
    const int* __restrict__ btok, const float* __restrict__ bw,
    const int* __restrict__ counts, const int* __restrict__ offs,
    int atomic_mode) {
  __shared__ bf16_t ld[2][64 * BPITCH];
  int e = blockIdx.z;
  int cnt = counts ? counts[e] : N_TOK;
  int rb = blockIdx.y * 128;
  if (rb >= cnt) return;
  int off = offs ? offs[e] : 0;
  const bf16_t* dwt = dT + (size_t)e * D_DIM * M_DIM;
  int lane = threadIdx.x & 31, wid = threadIdx.x >> 5;
  int wr = wid & 3, wc = wid >> 2;
  int half = lane >> 4, ln = lane & 15;
  int rowbase = rb + 32 * wr;
  int colblock = blockIdx.x * 64;
  int colbase = colblock + 32 * wc;

  const bf16_t* arow[2];
#pragma unroll
  for (int rs = 0; rs < 2; ++rs) {
    int r = rowbase + 16 * rs + ln;
    int rr = (r < cnt) ? r : (cnt - 1);
    arow[rs] = hbase + (size_t)(off + rr) * M_DIM;
  }

  // Copy assignment: wave wid handles rows wid*8 + lane>>2; 1 instr/step.
  int crow = wid * 8 + (lane >> 2);
  int cchunk = (lane & 3) * 8;
  const bf16_t* csrc = dwt + (size_t)(colblock + crow) * M_DIM + cchunk;
  bf16_t* cdst0 = ld[0] + crow * BPITCH + cchunk;
  bf16_t* cdst1 = ld[1] + crow * BPITCH + cchunk;

  v8f acc[2][2];
  v8f zv = {};
#pragma unroll
  for (int rs = 0; rs < 2; ++rs)
#pragma unroll
    for (int cs = 0; cs < 2; ++cs) acc[rs][cs] = zv;

  async_copy_b128(csrc, cdst0);
  int buf = 0;
  for (int kk = 0; kk < M_DIM; kk += 32) {
    if (kk + 32 < M_DIM) {
      async_copy_b128(csrc + kk + 32, buf ? cdst0 : cdst1);
      WAIT_ASYNC(1);
    } else {
      WAIT_ASYNC(0);
    }
    __syncthreads();

    int k0 = kk + half * 8;
    if (kk + 256 < M_DIM) __builtin_prefetch(arow[0] + kk + 256, 0, 1);
    const bf16_t* td = ld[buf];
    v16bf a[2], Bf[2];
#pragma unroll
    for (int rs = 0; rs < 2; ++rs) a[rs] = load_frag(arow[rs], k0);
#pragma unroll
    for (int cs = 0; cs < 2; ++cs)
      Bf[cs] = lds_frag(td, 32 * wc + 16 * cs + ln, half * 8);
#pragma unroll
    for (int rs = 0; rs < 2; ++rs)
#pragma unroll
      for (int cs = 0; cs < 2; ++cs)
        acc[rs][cs] = wmma_bf16(a[rs], Bf[cs], acc[rs][cs]);
    __syncthreads();
    buf ^= 1;
  }

#pragma unroll
  for (int rs = 0; rs < 2; ++rs)
#pragma unroll
    for (int cs = 0; cs < 2; ++cs) {
      int col = colbase + 16 * cs + ln;
#pragma unroll
      for (int j = 0; j < 8; ++j) {
        int row = rowbase + 16 * rs + j + 8 * half;
        if (row < cnt) {
          int rr = off + row;
          int tok = btok ? btok[rr] : row;
          float wv = bw ? bw[rr] : 1.f;
          float v = acc[rs][cs][j] * wv;
          float* p = &out[(size_t)tok * D_DIM + col];
          if (atomic_mode) atomicAdd(p, v);   // exactly 2 commutative adds/elem
          else *p += v;                        // shared expert, exclusive, last
        }
      }
    }
}

// ---------------------------------------------------------------- launch
extern "C" void kernel_launch(void* const* d_in, const int* in_sizes, int n_in,
                              void* d_out, int out_size, void* d_ws, size_t ws_size,
                              hipStream_t stream) {
  (void)in_sizes; (void)n_in; (void)out_size; (void)ws_size;
  const float* x   = (const float*)d_in[0];
  const float* U   = (const float*)d_in[1];
  const float* rw  = (const float*)d_in[2];
  const float* sb  = (const float*)d_in[3];
  const float* gb  = (const float*)d_in[4];
  const float* gw  = (const float*)d_in[5];
  const float* uw  = (const float*)d_in[6];
  const float* dw  = (const float*)d_in[7];
  const float* sgw = (const float*)d_in[8];
  const float* suw = (const float*)d_in[9];
  const float* sdw = (const float*)d_in[10];
  float* out = (float*)d_out;

  char* base = (char*)d_ws;
  size_t off = 0;
  auto carve = [&](size_t b) -> char* {
    char* p = base + off;
    off = (off + b + 255) & ~(size_t)255;
    return p;
  };
  bf16_t* xb   = (bf16_t*)carve((size_t)N_TOK * D_DIM * 2);
  bf16_t* gT   = (bf16_t*)carve((size_t)E_NUM * M_DIM * D_DIM * 2);
  bf16_t* uT   = (bf16_t*)carve((size_t)E_NUM * M_DIM * D_DIM * 2);
  bf16_t* dT   = (bf16_t*)carve((size_t)E_NUM * D_DIM * M_DIM * 2);
  bf16_t* sgT  = (bf16_t*)carve((size_t)M_DIM * D_DIM * 2);
  bf16_t* suT  = (bf16_t*)carve((size_t)M_DIM * D_DIM * 2);
  bf16_t* sdT  = (bf16_t*)carve((size_t)D_DIM * M_DIM * 2);
  bf16_t* h_rt = (bf16_t*)carve((size_t)2 * N_TOK * M_DIM * 2);
  bf16_t* h_sh = (bf16_t*)carve((size_t)N_TOK * M_DIM * 2);
  int*    tok_e = (int*)carve((size_t)2 * N_TOK * 4);
  float*  tok_w = (float*)carve((size_t)2 * N_TOK * 4);
  int*    btok  = (int*)carve((size_t)2 * N_TOK * 4);
  float*  bwt   = (float*)carve((size_t)2 * N_TOK * 4);
  int*    counts = (int*)carve(64);
  int*    offs   = (int*)carve(64);
  float*  partials = (float*)carve((size_t)(N_TOK / 8) * E_NUM * 4);

  dim3 tb(32, 8);
  // precision conversion + weight transposes (K-contiguous bf16)
  cvt_x_kernel<<<dim3((unsigned)(((size_t)N_TOK * D_DIM) / 1024)), 256, 0, stream>>>(x, xb);
  transpose_cvt<<<dim3(M_DIM / 32, D_DIM / 32, E_NUM), tb, 0, stream>>>(gw, gT, D_DIM, M_DIM);
  transpose_cvt<<<dim3(M_DIM / 32, D_DIM / 32, E_NUM), tb, 0, stream>>>(uw, uT, D_DIM, M_DIM);
  transpose_cvt<<<dim3(D_DIM / 32, M_DIM / 32, E_NUM), tb, 0, stream>>>(dw, dT, M_DIM, D_DIM);
  transpose_cvt<<<dim3(M_DIM / 32, D_DIM / 32, 1), tb, 0, stream>>>(sgw, sgT, D_DIM, M_DIM);
  transpose_cvt<<<dim3(M_DIM / 32, D_DIM / 32, 1), tb, 0, stream>>>(suw, suT, D_DIM, M_DIM);
  transpose_cvt<<<dim3(D_DIM / 32, M_DIM / 32, 1), tb, 0, stream>>>(sdw, sdT, M_DIM, D_DIM);

  // router + top-2 + bucket build
  router_kernel<<<N_TOK / 8, 256, 0, stream>>>(x, U, rw, sb, gb, tok_e, tok_w, partials);
  zero_counts_kernel<<<1, 64, 0, stream>>>(counts);
  count_kernel<<<N_TOK / 256, 256, 0, stream>>>(tok_e, counts);
  scan_kernel<<<1, 32, 0, stream>>>(counts, offs);
  place_kernel<<<E_NUM, 32, 0, stream>>>(tok_e, tok_w, offs, btok, bwt);

  // output init + bal loss (last element of d_out)
  zero_out_kernel<<<4096, 256, 0, stream>>>(out, (size_t)N_TOK * D_DIM + 1);
  balloss_kernel<<<1, 256, 0, stream>>>(partials, N_TOK / 8, out + (size_t)N_TOK * D_DIM);

  // grouped GEMMs: routed experts (gathered), then shared expert
  gemm_gateup<<<dim3(M_DIM / 64, N_TOK / 128, E_NUM), 256, 0, stream>>>(
      xb, gT, uT, h_rt, btok, counts, offs);
  gemm_gateup<<<dim3(M_DIM / 64, N_TOK / 128, 1), 256, 0, stream>>>(
      xb, sgT, suT, h_sh, nullptr, nullptr, nullptr);
  gemm_down<<<dim3(D_DIM / 64, N_TOK / 128, E_NUM), 256, 0, stream>>>(
      h_rt, dT, out, btok, bwt, counts, offs, 1);
  gemm_down<<<dim3(D_DIM / 64, N_TOK / 128, 1), 256, 0, stream>>>(
      h_sh, sdT, out, nullptr, nullptr, nullptr, nullptr, 0);
}
